// MultiModalGraphSAGE_5626407158207
// MI455X (gfx1250) — matrix operand order repaired and checked by reading, not compiled, not run on hardware
//
#include <hip/hip_runtime.h>
#include <hip/hip_fp16.h>

typedef __attribute__((ext_vector_type(16))) _Float16 v16h;
typedef __attribute__((ext_vector_type(8)))  float    v8f;

#define N_NODES 30000
#define N_EDGES 480000
#define DIM     512
#define FHID    300

// ---------------------------------------------------------------------------
// Aggregation kernels (bandwidth-bound, atomics path)
// ---------------------------------------------------------------------------
__global__ void deg_kernel(const int* __restrict__ dst, float* __restrict__ deg) {
    int e = blockIdx.x * blockDim.x + threadIdx.x;
    if (e < N_EDGES) atomicAdd(&deg[dst[e]], 1.0f);
}

__global__ void scatter_kernel(const float* __restrict__ x,
                               const int* __restrict__ src,
                               const int* __restrict__ dst,
                               float* __restrict__ agg) {
    long gid = (long)blockIdx.x * blockDim.x + threadIdx.x;
    const long total = (long)N_EDGES * (DIM / 4);
    if (gid >= total) return;
    int e = (int)(gid / (DIM / 4));
    int c = (int)(gid % (DIM / 4)) * 4;
    int s = src[e], d = dst[e];
    float4 v = *(const float4*)&x[(long)s * DIM + c];
    float* ap = &agg[(long)d * DIM + c];
    atomicAdd(ap + 0, v.x);
    atomicAdd(ap + 1, v.y);
    atomicAdd(ap + 2, v.z);
    atomicAdd(ap + 3, v.w);
}

__global__ void degnorm_kernel(float* __restrict__ agg, const float* __restrict__ deg) {
    long i = (long)blockIdx.x * blockDim.x + threadIdx.x;
    if (i < (long)N_NODES * DIM) {
        float d = fmaxf(deg[i / DIM], 1.0f);
        agg[i] = agg[i] / d;
    }
}

// ---------------------------------------------------------------------------
// WMMA GEMM:  C[M,Nout] = A0[M,K] @ B0[Nout,K]^T (+ A1 @ B1^T) + bias (+relu)
// Block tile 128x128, 8 waves (wave32), each wave 32x64 = 2x4 WMMA tiles.
// f32 global -> f16 register staging -> double-buffered LDS ->
// v_wmma_f32_16x16x32_f16.  OOB rows/cols are CLAMPED in the loader (their
// accumulator lanes are never stored); only the K-tail needs zero-fill (one
// float4 predicate; all K are multiples of 4).
// ---------------------------------------------------------------------------
#define BM 128
#define BN 128
#define KT 32
#define LDA 40   // halfs per LDS row (80B: 16B-aligned fragments, padded)

union H4 { _Float16 h[4]; unsigned long long u; };

__device__ __forceinline__ void load_tile_regs(
    const float* __restrict__ A, const float* __restrict__ B,
    int blockM, int blockN, int M, int K, int Nout,
    int k0, int lr, int lc, H4* ra, H4* rb)
{
    const int gk = k0 + lc;
    const bool kin = (gk + 3) < K;          // K % 4 == 0 for all call sites
#pragma unroll
    for (int p = 0; p < 4; ++p) {
        int row = p * 32 + lr;
        int gm = blockM + row; gm = gm < M    ? gm : M - 1;     // clamp: masked at store
        int gn = blockN + row; gn = gn < Nout ? gn : Nout - 1;  // clamp: masked at store
        float4 va = make_float4(0.f, 0.f, 0.f, 0.f);
        float4 vb = va;
        if (kin) {
            va = *(const float4*)&A[(long)gm * K + gk];
            vb = *(const float4*)&B[(long)gn * K + gk];
        }
        ra[p].h[0] = (_Float16)va.x; ra[p].h[1] = (_Float16)va.y;
        ra[p].h[2] = (_Float16)va.z; ra[p].h[3] = (_Float16)va.w;
        rb[p].h[0] = (_Float16)vb.x; rb[p].h[1] = (_Float16)vb.y;
        rb[p].h[2] = (_Float16)vb.z; rb[p].h[3] = (_Float16)vb.w;
    }
}

__device__ __forceinline__ void store_tile_lds(
    _Float16* __restrict__ As, _Float16* __restrict__ Bs,
    int lr, int lc, const H4* ra, const H4* rb)
{
#pragma unroll
    for (int p = 0; p < 4; ++p) {
        int row = p * 32 + lr;
        *(unsigned long long*)&As[row * LDA + lc] = ra[p].u;
        *(unsigned long long*)&Bs[row * LDA + lc] = rb[p].u;
    }
}

__global__ __launch_bounds__(256) void wmma_gemm_kernel(
    const float* __restrict__ A0, const float* __restrict__ B0,
    const float* __restrict__ A1, const float* __restrict__ B1,
    const float* __restrict__ bias, float* __restrict__ C,
    int M, int K, int Nout, int relu, int npass)
{
    __shared__ _Float16 As[2][BM * LDA];
    __shared__ _Float16 Bs[2][BN * LDA];

    const int tid   = threadIdx.x;
    const int lane  = tid & 31;
    const int w     = tid >> 5;
    const int laneM = lane & 15;
    const int laneH = lane >> 4;
    const int wm    = (w & 3) * 32;   // 4 waves along M
    const int wn    = (w >> 2) * 64;  // 2 waves along N
    const int blockN = blockIdx.x * BN;
    const int blockM = blockIdx.y * BM;

    const int kChunks = (K + KT - 1) / KT;
    const int nIter   = kChunks * npass;

    // tile loader indexing: 8 threads/row * float4 -> 32 rows/pass, 4 passes
    const int lr = tid >> 3;
    const int lc = (tid & 7) * 4;

    v8f acc[2][4] = {};
    H4 ra[4], rb[4];

    // prologue: stage + store tile 0
    load_tile_regs(A0, B0, blockM, blockN, M, K, Nout, 0, lr, lc, ra, rb);
    store_tile_lds(As[0], Bs[0], lr, lc, ra, rb);
    __syncthreads();

    int pass = 0, k0 = 0;   // coordinates of the tile resident in buffer (it&1)
    for (int it = 0; it < nIter; ++it) {
        const int cur = it & 1;
        // next tile coordinates (no division)
        int nk = k0 + KT, np = pass;
        if (nk >= K) { nk = 0; np = pass + 1; }
        const bool more = (it + 1) < nIter;

        if (more)   // issue global loads for tile it+1; overlap with WMMAs below
            load_tile_regs(np ? A1 : A0, np ? B1 : B0,
                           blockM, blockN, M, K, Nout, nk, lr, lc, ra, rb);

        // Fragment fetch per CDNA5 16-bit layouts:
        // A 16x32: lane half picks K halves {0..7,16..23} vs {8..15,24..31}
        // B 32x16: lane = column; lane half picks K halves {0..7,16..23} vs {8..15,24..31}
        union F16 { v16h v; uint4 q[2]; };
        F16 af[2], bf[4];
#pragma unroll
        for (int i = 0; i < 2; ++i) {
            int off = (wm + i * 16 + laneM) * LDA + laneH * 8;
            af[i].q[0] = *(const uint4*)&As[cur][off];
            af[i].q[1] = *(const uint4*)&As[cur][off + 16];
        }
#pragma unroll
        for (int j = 0; j < 4; ++j) {
            int off = (wn + j * 16 + laneM) * LDA + laneH * 8;
            bf[j].q[0] = *(const uint4*)&Bs[cur][off];
            bf[j].q[1] = *(const uint4*)&Bs[cur][off + 16];
        }
#pragma unroll
        for (int i = 0; i < 2; ++i)
#pragma unroll
            for (int j = 0; j < 4; ++j)
                acc[i][j] = __builtin_amdgcn_wmma_f32_16x16x32_f16(
                    false, af[i].v, false, bf[j].v,
                    (short)0, acc[i][j], false, false);

        if (more)   // park staged tile into the other buffer
            store_tile_lds(As[1 - cur], Bs[1 - cur], lr, lc, ra, rb);
        __syncthreads();

        k0 = nk; pass = np;
    }

    // -----------------------------------------------------------------------
    // Epilogue.  C/D layout: lane&15 = N col; VGPR r -> M = r + (lane>>4)*8.
    // Bias hoisted (clamped index); row addressing strength-reduced (+Nout
    // per r, no 64-bit multiplies); uniform fast path for interior tiles.
    // -----------------------------------------------------------------------
    const int colBase = blockN + wn + laneM;
    float bv[4];
#pragma unroll
    for (int j = 0; j < 4; ++j) {
        int gn = colBase + j * 16;
        bv[j] = bias[gn < Nout ? gn : Nout - 1];
    }

    const bool fullTile = (blockM + BM <= M) && (blockN + BN <= Nout);
    if (fullTile) {
#pragma unroll
        for (int i = 0; i < 2; ++i) {
            const int row0 = blockM + wm + i * 16 + laneH * 8;
            float* rowPtr = C + (long)row0 * Nout + colBase;
#pragma unroll
            for (int r = 0; r < 8; ++r) {
#pragma unroll
                for (int j = 0; j < 4; ++j) {
                    float v = acc[i][j][r] + bv[j];
                    if (relu) v = fmaxf(v, 0.f);
                    rowPtr[j * 16] = v;
                }
                rowPtr += Nout;
            }
        }
    } else {
#pragma unroll
        for (int i = 0; i < 2; ++i) {
            const int row0 = blockM + wm + i * 16 + laneH * 8;
            float* rowPtr = C + (long)row0 * Nout + colBase;
#pragma unroll
            for (int r = 0; r < 8; ++r) {
                const bool mOk = (row0 + r) < M;
#pragma unroll
                for (int j = 0; j < 4; ++j) {
                    if (mOk && (colBase + j * 16) < Nout) {
                        float v = acc[i][j][r] + bv[j];
                        if (relu) v = fmaxf(v, 0.f);
                        rowPtr[j * 16] = v;
                    }
                }
                rowPtr += Nout;
            }
        }
    }
}

// ---------------------------------------------------------------------------
// Gate softmax + weighted fusion + L2 normalize.  gph lives in the `fused`
// output slot and is overwritten in place (read-before-write per wave/node).
// One wave (32 lanes) per node; 8 waves per block; 3750*8 == 30000 exactly.
// ---------------------------------------------------------------------------
__global__ __launch_bounds__(256) void fuse_kernel(
    float* __restrict__ gph_fused, const float* __restrict__ img,
    const float* __restrict__ attr, const float* __restrict__ rel,
    const float* __restrict__ Wfp, const float* __restrict__ bfp)
{
    __shared__ float sW[FHID];
    __shared__ float sF[8][FHID];
    int tid = threadIdx.x;
    for (int f = tid; f < FHID; f += 256) sW[f] = Wfp[f];
    __syncthreads();

    int w = tid >> 5, lane = tid & 31;
    int node = blockIdx.x * 8 + w;   // exact: no guard needed

    const float* gp = gph_fused + (long)node * FHID;
    const float* ip = img       + (long)node * FHID;
    const float* ap = attr      + (long)node * FHID;
    const float* rp = rel       + (long)node * FHID;

    float g0 = 0.f, g1 = 0.f, g2 = 0.f, g3 = 0.f;
    for (int f = lane; f < FHID; f += 32) {
        float wv = sW[f];
        g0 += gp[f] * wv; g1 += ip[f] * wv;
        g2 += ap[f] * wv; g3 += rp[f] * wv;
    }
    for (int m = 16; m; m >>= 1) {
        g0 += __shfl_xor(g0, m, 32);
        g1 += __shfl_xor(g1, m, 32);
        g2 += __shfl_xor(g2, m, 32);
        g3 += __shfl_xor(g3, m, 32);
    }
    float bb = bfp[0];
    g0 += bb; g1 += bb; g2 += bb; g3 += bb;

    float mx = fmaxf(fmaxf(g0, g1), fmaxf(g2, g3));
    float e0 = expf(g0 - mx), e1 = expf(g1 - mx);
    float e2 = expf(g2 - mx), e3 = expf(g3 - mx);
    float inv_es = 1.0f / (e0 + e1 + e2 + e3);
    float w0 = e0 * inv_es, w1 = e1 * inv_es, w2 = e2 * inv_es, w3 = e3 * inv_es;

    float nrm = 0.f;
    for (int f = lane; f < FHID; f += 32) {
        float v = w0 * gp[f] + w1 * ip[f] + w2 * ap[f] + w3 * rp[f];
        sF[w][f] = v;
        nrm += v * v;
    }
    for (int m = 16; m; m >>= 1) nrm += __shfl_xor(nrm, m, 32);
    float inv = 1.0f / fmaxf(sqrtf(nrm), 1e-12f);

    float* op = gph_fused + (long)node * FHID;
    for (int f = lane; f < FHID; f += 32) op[f] = sF[w][f] * inv;
}

// ---------------------------------------------------------------------------
// Launch
// ---------------------------------------------------------------------------
extern "C" void kernel_launch(void* const* d_in, const int* in_sizes, int n_in,
                              void* d_out, int out_size, void* d_ws, size_t ws_size,
                              hipStream_t stream) {
    const float* x        = (const float*)d_in[0];
    const int*   ei       = (const int*)  d_in[1];
    const float* img_emb  = (const float*)d_in[2];
    const float* attr_emb = (const float*)d_in[3];
    const float* rel_emb  = (const float*)d_in[4];
    const float* Wl1 = (const float*)d_in[5],  *Wr1 = (const float*)d_in[6],  *b1 = (const float*)d_in[7];
    const float* Wl2 = (const float*)d_in[8],  *Wr2 = (const float*)d_in[9],  *b2 = (const float*)d_in[10];
    const float* Wl3 = (const float*)d_in[11], *Wr3 = (const float*)d_in[12], *b3 = (const float*)d_in[13];
    const float* Wg  = (const float*)d_in[14], *bg  = (const float*)d_in[15];
    const float* Wi  = (const float*)d_in[16], *bi  = (const float*)d_in[17];
    const float* Wa  = (const float*)d_in[18], *ba  = (const float*)d_in[19];
    const float* Wrl = (const float*)d_in[20], *brl = (const float*)d_in[21];
    const float* Wfp = (const float*)d_in[22], *bfp = (const float*)d_in[23];

    const int* src = ei;
    const int* dst = ei + N_EDGES;

    // outputs: fused[N,300], h[N,512], img[N,300], attr[N,300], rel[N,300]
    float* out    = (float*)d_out;
    float* fusedO = out;                               // also holds gph pre-fusion
    float* hO     = out + (size_t)N_NODES * 300;
    float* imgO   = out + (size_t)N_NODES * 812;
    float* attrO  = out + (size_t)N_NODES * 1112;
    float* relO   = out + (size_t)N_NODES * 1412;

    // workspace: deg | agg | buf0 | buf1
    float* ws   = (float*)d_ws;
    float* deg  = ws;
    float* agg  = ws + 30720;
    float* buf0 = agg  + (size_t)N_NODES * DIM;
    float* buf1 = buf0 + (size_t)N_NODES * DIM;

    const size_t aggBytes = (size_t)N_NODES * DIM * sizeof(float);
    const long scatterThreads = (long)N_EDGES * (DIM / 4);
    const int  scatterBlocks  = (int)((scatterThreads + 255) / 256);
    const int  normBlocks     = (int)(((long)N_NODES * DIM + 255) / 256);

    dim3 blk(256);
    dim3 gS((512 + BN - 1) / BN, (N_NODES + BM - 1) / BM);   // (4, 235)
    dim3 gF((FHID + BN - 1) / BN, (N_NODES + BM - 1) / BM);  // (3, 235)

    // degree (only depends on dst -> compute once)
    hipMemsetAsync(deg, 0, 30720 * sizeof(float), stream);
    deg_kernel<<<(N_EDGES + 255) / 256, blk, 0, stream>>>(dst, deg);

    // ---- SAGE layer 1 ----
    hipMemsetAsync(agg, 0, aggBytes, stream);
    scatter_kernel<<<scatterBlocks, blk, 0, stream>>>(x, src, dst, agg);
    degnorm_kernel<<<normBlocks, blk, 0, stream>>>(agg, deg);
    wmma_gemm_kernel<<<gS, blk, 0, stream>>>(agg, Wl1, x, Wr1, b1, buf0,
                                             N_NODES, DIM, DIM, 1, 2);
    // ---- SAGE layer 2 ----
    hipMemsetAsync(agg, 0, aggBytes, stream);
    scatter_kernel<<<scatterBlocks, blk, 0, stream>>>(buf0, src, dst, agg);
    degnorm_kernel<<<normBlocks, blk, 0, stream>>>(agg, deg);
    wmma_gemm_kernel<<<gS, blk, 0, stream>>>(agg, Wl2, buf0, Wr2, b2, buf1,
                                             N_NODES, DIM, DIM, 1, 2);
    // ---- SAGE layer 3 (h -> output slot, no relu) ----
    hipMemsetAsync(agg, 0, aggBytes, stream);
    scatter_kernel<<<scatterBlocks, blk, 0, stream>>>(buf1, src, dst, agg);
    degnorm_kernel<<<normBlocks, blk, 0, stream>>>(agg, deg);
    wmma_gemm_kernel<<<gS, blk, 0, stream>>>(agg, Wl3, buf1, Wr3, b3, hO,
                                             N_NODES, DIM, DIM, 0, 2);

    // ---- fusion projections ----
    wmma_gemm_kernel<<<gF, blk, 0, stream>>>(hO, Wg, nullptr, nullptr, bg, fusedO,
                                             N_NODES, DIM, FHID, 0, 1);      // gph
    wmma_gemm_kernel<<<gF, blk, 0, stream>>>(img_emb, Wi, nullptr, nullptr, bi, imgO,
                                             N_NODES, 2048, FHID, 0, 1);
    wmma_gemm_kernel<<<gF, blk, 0, stream>>>(attr_emb, Wa, nullptr, nullptr, ba, attrO,
                                             N_NODES, 1000, FHID, 0, 1);
    wmma_gemm_kernel<<<gF, blk, 0, stream>>>(rel_emb, Wrl, nullptr, nullptr, brl, relO,
                                             N_NODES, 1000, FHID, 0, 1);

    // ---- gate softmax + fuse + L2 normalize (gph -> fused, in place) ----
    fuse_kernel<<<N_NODES / 8, blk, 0, stream>>>(fusedO, imgO, attrO, relO, Wfp, bfp);
}
